// ARC_net_33509334843929
// MI455X (gfx1250) — matrix operand order repaired and checked by reading, not compiled, not run on hardware
//
#include <hip/hip_runtime.h>

// ---------------------------------------------------------------------------
// MI455X / gfx1250 implementation.
// GEMMs: v_wmma_f32_16x16x32_f16, 1x4 tile blocking per wave, B panel staged
// in LDS via gfx1250 async loads (global_load_async_to_lds_b128 +
// s_wait_asynccnt). Data is L2-resident (~33MB << 192MB L2) -> matrix-pipe
// bound, not HBM bound.
// ---------------------------------------------------------------------------

#define R_     2048
#define A_     16384
#define E_     131072
#define KNBR_  48
#define RAD2_  100.0f     // RADIUS^2
#define KC_    128        // k-chunk (halves) staged in LDS per round

typedef __attribute__((ext_vector_type(16))) _Float16 v16h;
typedef __attribute__((ext_vector_type(8)))  float    v8f;

// ----------------------------- small utility kernels -----------------------

__global__ void zero_f32(float* p, int n) {
    int i = blockIdx.x * blockDim.x + threadIdx.x;
    if (i < n) p[i] = 0.0f;
}

// src: f32 [K,N] row-major  ->  dst: f16 [N, ldk] row-major at k-offset koff
__global__ void convtrans(const float* __restrict__ src, _Float16* __restrict__ dst,
                          int K, int N, int ldk, int koff) {
    int i = blockIdx.x * blockDim.x + threadIdx.x;
    if (i >= K * N) return;
    int k = i / N, n = i % N;
    dst[(size_t)n * ldk + koff + k] = (_Float16)src[(size_t)k * N + n];
}

// res_feat = node_feature @ Wres + bres   (tiny: 21 MACs/output, scalar VALU)
__global__ void resfeat_kernel(const float* __restrict__ nf, const float* __restrict__ Wres,
                               const float* __restrict__ bres, float* __restrict__ out) {
    int i = blockIdx.x * blockDim.x + threadIdx.x;
    if (i >= R_ * 64) return;
    int r = i >> 6, c = i & 63;
    float s = bres[c];
    const float* row = nf + (size_t)r * 21;
#pragma unroll
    for (int j = 0; j < 21; ++j) s += row[j] * Wres[j * 64 + c];
    out[i] = s;
}

// ----------------------------- ball query (top-48) --------------------------
// One 256-thread workgroup per residue. Tournament selection:
// per-thread chunk-of-64 local min in LDS, 48 rounds of wave32 xor-reduce +
// rescan of only the winning chunk (selected atoms in a 2KB LDS bitmask).
__global__ __launch_bounds__(256)
void ballquery_kernel(const float* __restrict__ pos, const int* __restrict__ ca_idx,
                      int* __restrict__ nbr, float* __restrict__ dist2) {
    __shared__ float lval[256];
    __shared__ int   lidx[256];
    __shared__ unsigned int sel[A_ / 32];   // 512 words = 16384 bits
    __shared__ int   swin[1];

    const int r = blockIdx.x;
    const int t = threadIdx.x;

    sel[t] = 0u;
    sel[t + 256] = 0u;

    const int ca = ca_idx[r];
    const float cx = pos[ca * 3 + 0], cy = pos[ca * 3 + 1], cz = pos[ca * 3 + 2];

    float bv = 3.0e38f; int bi = -1;
#pragma unroll 4
    for (int j = 0; j < 64; ++j) {
        int a = t * 64 + j;
        float dx = pos[a * 3 + 0] - cx;
        float dy = pos[a * 3 + 1] - cy;
        float dz = pos[a * 3 + 2] - cz;
        float d2 = dx * dx + dy * dy + dz * dz;
        if (d2 < bv) { bv = d2; bi = a; }
    }
    lval[t] = bv; lidx[t] = bi;
    __syncthreads();

    for (int it = 0; it < KNBR_; ++it) {
        if (t < 32) {
            float v = 3.0e38f; int i = -1;
#pragma unroll
            for (int j = 0; j < 8; ++j) {
                int c = t * 8 + j;
                float cv = lval[c];
                if (cv < v) { v = cv; i = lidx[c]; }
            }
#pragma unroll
            for (int off = 16; off > 0; off >>= 1) {   // wave32 xor-reduce
                float ov = __shfl_xor(v, off, 32);
                int   oi = __shfl_xor(i, off, 32);
                if (ov < v) { v = ov; i = oi; }
            }
            if (t == 0) {
                nbr[r * KNBR_ + it]   = i;
                dist2[r * KNBR_ + it] = v;
                sel[i >> 5] |= (1u << (i & 31));
                swin[0] = i;
            }
        }
        __syncthreads();
        const int w = swin[0];
        if (t == (w >> 6)) {
            float v = 3.0e38f; int i = -1;
            for (int j = 0; j < 64; ++j) {
                int a = t * 64 + j;
                if ((sel[a >> 5] >> (a & 31)) & 1u) continue;
                float dx = pos[a * 3 + 0] - cx;
                float dy = pos[a * 3 + 1] - cy;
                float dz = pos[a * 3 + 2] - cz;
                float d2 = dx * dx + dy * dy + dz * dz;
                if (d2 < v) { v = d2; i = a; }
            }
            lval[t] = v; lidx[t] = i;
        }
        __syncthreads();
    }
}

// pts[r,k,:] = mask * (emb_atom[atom_name[nbr]] + delta @ Wxyz)  -> f16
__global__ void buildpts_kernel(const float* __restrict__ pos, const int* __restrict__ ca_idx,
                                const int* __restrict__ atom_name, const int* __restrict__ nbr,
                                const float* __restrict__ dist2, const float* __restrict__ emb,
                                const float* __restrict__ Wxyz, _Float16* __restrict__ ptsA) {
    int i = blockIdx.x * blockDim.x + threadIdx.x;    // i = r*48 + k
    if (i >= R_ * KNBR_) return;
    int r = i / KNBR_;
    int a = nbr[i];
    float m = (dist2[i] < RAD2_) ? 1.0f : 0.0f;
    int ca = ca_idx[r];
    float dx = pos[a * 3 + 0] - pos[ca * 3 + 0];
    float dy = pos[a * 3 + 1] - pos[ca * 3 + 1];
    float dz = pos[a * 3 + 2] - pos[ca * 3 + 2];
    const float* e = emb + (size_t)atom_name[a] * 64;
    _Float16* o = ptsA + (size_t)i * 64;
#pragma unroll 8
    for (int c = 0; c < 64; ++c) {
        float v = e[c] + dx * Wxyz[c] + dy * Wxyz[64 + c] + dz * Wxyz[128 + c];
        o[c] = (_Float16)(v * m);
    }
}

// relational scatter-sum: agg[dst, rel*64 + c] += res_feat[src, c]
__global__ void scatter_kernel(const int* __restrict__ esrc, const int* __restrict__ edst,
                               const int* __restrict__ erel, const float* __restrict__ rf,
                               float* __restrict__ agg) {
    int e = blockIdx.x * blockDim.x + threadIdx.x;
    if (e >= E_) return;
    const float* f = rf + (size_t)esrc[e] * 64;
    float* o = agg + (size_t)edst[e] * 448 + (size_t)erel[e] * 64;
#pragma unroll 8
    for (int c = 0; c < 64; ++c) atomicAdd(o + c, f[c]);
}

// A1 = [agg | res_feat] as f16  [2048, 512]
__global__ void packA1_kernel(const float* __restrict__ agg, const float* __restrict__ rf,
                              _Float16* __restrict__ A1) {
    int i = blockIdx.x * blockDim.x + threadIdx.x;
    if (i >= R_ * 512) return;
    int r = i >> 9, c = i & 511;
    float v = (c < 448) ? agg[(size_t)r * 448 + c] : rf[(size_t)r * 64 + (c - 448)];
    A1[i] = (_Float16)v;
}

__global__ void f32tof16_kernel(const float* __restrict__ src, _Float16* __restrict__ dst, int n) {
    int i = blockIdx.x * blockDim.x + threadIdx.x;
    if (i < n) dst[i] = (_Float16)src[i];
}

// ----------------------------- generic WMMA GEMM ----------------------------
// A: f16 [M,K] row-major (lda == K). Bt: f16 [N,K] row-major (pre-transposed).
// Each wave: 1 M-tile x 4 N-tiles (one A-frag feeds 4 WMMAs). The 64-column
// B panel is shared by the block: staged in LDS with gfx1250 async loads,
// fenced by s_wait_asynccnt + barrier, read back as ds_load_b128 fragments.
// Requires: M % 128 == 0, N % 64 == 0, K % 32 == 0 (all call sites satisfy).

__device__ inline v16h load_a_frag(const _Float16* __restrict__ A, int K,
                                   int mt, int ktg, int lane) {
    int mr = lane & 15, half = lane >> 4;
    const _Float16* row = A + (size_t)(mt * 16 + mr) * K + ktg * 32;
    int kb0 = half ? 16 : 0;      // VGPRs 0-3
    int kb1 = half ? 24 : 8;      // VGPRs 4-7
    v16h f;
#pragma unroll
    for (int v = 0; v < 8; ++v) f[v] = row[kb0 + v];
#pragma unroll
    for (int v = 0; v < 8; ++v) f[8 + v] = row[kb1 + v];
    return f;
}

__device__ inline v16h load_b_frag_lds(const _Float16* __restrict__ sB, int kc,
                                       int j, int kt, int nr, int half) {
    const _Float16* p = sB + (size_t)(j * 16 + nr) * kc + kt * 32 + (half ? 16 : 0);
    v16h f;
#pragma unroll
    for (int v = 0; v < 16; ++v) f[v] = p[v];   // -> 2x ds_load_b128
    return f;
}

__global__ __launch_bounds__(256)
void gemm_wmma(const _Float16* __restrict__ A, const _Float16* __restrict__ Bt,
               const float* __restrict__ bias,
               float* __restrict__ Cf32, int ldc,
               _Float16* __restrict__ Cf16, int ldh,
               int M, int N, int K, int relu,
               float* __restrict__ pool, const float* __restrict__ dist2) {
    __shared__ _Float16 sB[64 * KC_];           // 16 KB B panel chunk

    const int lane    = threadIdx.x & 31;
    const int wave    = threadIdx.x >> 5;
    const int ngroups = N >> 6;                 // groups of 4 N-tiles
    const int ngroup  = blockIdx.x % ngroups;
    const int blkm    = blockIdx.x / ngroups;
    const int mt      = blkm * 8 + wave;        // one M-tile per wave
    const int nbase   = ngroup * 64;
    const int nr      = lane & 15, half = lane >> 4;

    const v8f vzero = {0.f, 0.f, 0.f, 0.f, 0.f, 0.f, 0.f, 0.f};
    v8f acc[4];
#pragma unroll
    for (int j = 0; j < 4; ++j) acc[j] = vzero;

    const int kc = (K < KC_) ? K : KC_;
    // low 32 bits of the flat shared address == LDS byte offset (ISA flat->LDS)
    const unsigned sbase = (unsigned)(size_t)(const void*)sB;

    for (int k0 = 0; k0 < K; k0 += kc) {
        __syncthreads();                        // protect previous chunk readers
        // ---- async-stage Bt[nbase:nbase+64, k0:k0+kc) into LDS --------------
        const int n128 = (64 * kc) >> 3;        // 16B packets (multiple of 256)
        for (int i = threadIdx.x; i < n128; i += 256) {
            int h0 = i << 3;
            int row = h0 / kc, col = h0 % kc;
            const _Float16* g = Bt + (size_t)(nbase + row) * K + k0 + col;
            unsigned lds = sbase + (unsigned)(h0 * 2);
            asm volatile("global_load_async_to_lds_b128 %0, %1, off"
                         :: "v"(lds), "v"(g) : "memory");
        }
        asm volatile("s_wait_asynccnt 0x0" ::: "memory");
        __syncthreads();
        // ---- compute: 4 WMMAs per A-fragment --------------------------------
        const int ksteps = kc >> 5;
        for (int kt = 0; kt < ksteps; ++kt) {
            v16h a = load_a_frag(A, K, mt, (k0 >> 5) + kt, lane);
#pragma unroll
            for (int j = 0; j < 4; ++j) {
                v16h b = load_b_frag_lds(sB, kc, j, kt, nr, half);
                acc[j] = __builtin_amdgcn_wmma_f32_16x16x32_f16(
                    false, a, false, b, (short)0, acc[j], false, false);
            }
        }
    }

    const int mbase = mt * 16 + (half ? 8 : 0);
#pragma unroll
    for (int j = 0; j < 4; ++j) {
        const int n = nbase + j * 16 + nr;
        const float bv = bias ? bias[n] : 0.0f;
        if (pool) {
            // fused masked max-pool: 48 rows/residue == 3 tiles, never straddles.
            // relu output >= 0 -> int atomicMax on float bits is exact.
#pragma unroll
            for (int v = 0; v < 8; ++v) {
                int m = mbase + v;
                float x = acc[j][v] + bv;
                x = x > 0.f ? x : 0.f;
                if (dist2[m] < RAD2_)
                    atomicMax((int*)&pool[(size_t)(m / KNBR_) * N + n], __float_as_int(x));
            }
        } else {
#pragma unroll
            for (int v = 0; v < 8; ++v) {
                int m = mbase + v;
                float x = acc[j][v] + bv;
                if (relu) x = x > 0.f ? x : 0.f;
                if (Cf32) Cf32[(size_t)m * ldc + n] = x;
                if (Cf16) Cf16[(size_t)m * ldh + n] = (_Float16)x;
            }
        }
    }
}

// ----------------------------- batch norm -----------------------------------

__global__ void bn_stats(const float* __restrict__ z, float* __restrict__ bnm,
                         float* __restrict__ bnr) {
    __shared__ float ss[256], ss2[256];
    const int c = blockIdx.x;
    float s = 0.f, s2 = 0.f;
    for (int r = threadIdx.x; r < R_; r += 256) {
        float v = z[(size_t)r * 512 + c];
        s += v; s2 += v * v;
    }
    ss[threadIdx.x] = s; ss2[threadIdx.x] = s2;
    __syncthreads();
    for (int o = 128; o > 0; o >>= 1) {
        if (threadIdx.x < o) {
            ss[threadIdx.x] += ss[threadIdx.x + o];
            ss2[threadIdx.x] += ss2[threadIdx.x + o];
        }
        __syncthreads();
    }
    if (threadIdx.x == 0) {
        float mean = ss[0] * (1.0f / R_);
        float var = ss2[0] * (1.0f / R_) - mean * mean;
        bnm[c] = mean;
        bnr[c] = rsqrtf(var + 1e-5f);
    }
}

__global__ void bn_apply(const float* __restrict__ z, const float* __restrict__ bnm,
                         const float* __restrict__ bnr, const float* __restrict__ gamma,
                         const float* __restrict__ beta, _Float16* __restrict__ A3) {
    int i = blockIdx.x * blockDim.x + threadIdx.x;
    if (i >= R_ * 512) return;
    int c = i & 511;
    float v = (z[i] - bnm[c]) * bnr[c] * gamma[c] + beta[c];
    v = v > 0.f ? v : 0.f;
    A3[i] = (_Float16)v;
}

// ----------------------------- launcher -------------------------------------

extern "C" void kernel_launch(void* const* d_in, const int* in_sizes, int n_in,
                              void* d_out, int out_size, void* d_ws, size_t ws_size,
                              hipStream_t stream) {
    const float* node_feature = (const float*)d_in[0];
    const float* pos          = (const float*)d_in[1];
    const int*   atom_name    = (const int*)d_in[2];
    const int*   ca_idx       = (const int*)d_in[3];
    const int*   edge_src     = (const int*)d_in[4];
    const int*   edge_dst     = (const int*)d_in[5];
    const int*   edge_rel     = (const int*)d_in[6];
    const float* emb_atom     = (const float*)d_in[7];
    const float* Wres         = (const float*)d_in[8];
    const float* bres         = (const float*)d_in[9];
    const float* Wxyz         = (const float*)d_in[10];
    const float* W1           = (const float*)d_in[11];
    const float* b1           = (const float*)d_in[12];
    const float* W2           = (const float*)d_in[13];
    const float* b2           = (const float*)d_in[14];
    const float* Wconv        = (const float*)d_in[15];
    const float* bconv        = (const float*)d_in[16];
    const float* Wself        = (const float*)d_in[17];
    const float* We1          = (const float*)d_in[18];
    const float* be1          = (const float*)d_in[19];
    const float* gamma1       = (const float*)d_in[20];
    const float* beta1        = (const float*)d_in[21];
    const float* We2          = (const float*)d_in[22];
    const float* be2          = (const float*)d_in[23];

    // -------- workspace layout (256B aligned), total ~33 MB ----------------
    char* ws = (char*)d_ws;
    size_t off = 0;
    auto alloc = [&](size_t bytes) { size_t c = off; off += (bytes + 255) & ~(size_t)255; return c; };
    int*       nbr    = (int*)      (ws + alloc((size_t)R_ * KNBR_ * 4));
    float*     dist2  = (float*)    (ws + alloc((size_t)R_ * KNBR_ * 4));
    _Float16*  ptsA   = (_Float16*) (ws + alloc((size_t)R_ * KNBR_ * 64 * 2));
    float*     envmax = (float*)    (ws + alloc((size_t)R_ * 128 * 4));
    _Float16*  envA   = (_Float16*) (ws + alloc((size_t)R_ * 128 * 2));
    float*     rfeat  = (float*)    (ws + alloc((size_t)R_ * 64 * 4));
    float*     agg    = (float*)    (ws + alloc((size_t)R_ * 448 * 4));
    _Float16*  A1     = (_Float16*) (ws + alloc((size_t)R_ * 512 * 2));
    _Float16*  A2     = (_Float16*) (ws + alloc((size_t)R_ * 768 * 2));
    float*     z      = (float*)    (ws + alloc((size_t)R_ * 512 * 4));
    float*     bnm    = (float*)    (ws + alloc(512 * 4));
    float*     bnr    = (float*)    (ws + alloc(512 * 4));
    _Float16*  A3     = (_Float16*) (ws + alloc((size_t)R_ * 512 * 2));
    _Float16*  W1t    = (_Float16*) (ws + alloc((size_t)128 * 64 * 2));
    _Float16*  W2t    = (_Float16*) (ws + alloc((size_t)256 * 128 * 2));
    _Float16*  Wcs    = (_Float16*) (ws + alloc((size_t)512 * 512 * 2));
    _Float16*  We1t   = (_Float16*) (ws + alloc((size_t)512 * 768 * 2));
    _Float16*  We2t   = (_Float16*) (ws + alloc((size_t)512 * 512 * 2));

    auto blocks = [](size_t n) { return dim3((unsigned)((n + 255) / 256)); };
    auto ggrid  = [](int M, int N) { return dim3((unsigned)((M / 128) * (N / 64))); };
    const dim3 T(256);

    // zero accumulators (ws is poisoned; must re-zero every call)
    zero_f32<<<blocks((size_t)R_ * 128), T, 0, stream>>>(envmax, R_ * 128);
    zero_f32<<<blocks((size_t)R_ * 448), T, 0, stream>>>(agg, R_ * 448);

    // pre-transpose weights to f16 [N,K]
    convtrans<<<blocks((size_t)64 * 128), T, 0, stream>>>(W1, W1t, 64, 128, 64, 0);
    convtrans<<<blocks((size_t)128 * 256), T, 0, stream>>>(W2, W2t, 128, 256, 128, 0);
    convtrans<<<blocks((size_t)448 * 512), T, 0, stream>>>(Wconv, Wcs, 448, 512, 512, 0);
    convtrans<<<blocks((size_t)64 * 512), T, 0, stream>>>(Wself, Wcs, 64, 512, 512, 448);
    convtrans<<<blocks((size_t)768 * 512), T, 0, stream>>>(We1, We1t, 768, 512, 768, 0);
    convtrans<<<blocks((size_t)512 * 512), T, 0, stream>>>(We2, We2t, 512, 512, 512, 0);

    // residue features + ball query + point features
    resfeat_kernel<<<blocks((size_t)R_ * 64), T, 0, stream>>>(node_feature, Wres, bres, rfeat);
    ballquery_kernel<<<dim3(R_), T, 0, stream>>>(pos, ca_idx, nbr, dist2);
    buildpts_kernel<<<blocks((size_t)R_ * KNBR_), T, 0, stream>>>(
        pos, ca_idx, atom_name, nbr, dist2, emb_atom, Wxyz, ptsA);

    // point MLP GEMM with fused masked max-pool: [98304,64]x[64,128]
    gemm_wmma<<<ggrid(R_ * KNBR_, 128), T, 0, stream>>>(
        ptsA, W1t, b1, nullptr, 0, nullptr, 0,
        R_ * KNBR_, 128, 64, 1, envmax, dist2);

    // env = relu(envmax @ W2 + b2) -> f16 into A2[:, 512:768]
    f32tof16_kernel<<<blocks((size_t)R_ * 128), T, 0, stream>>>(envmax, envA, R_ * 128);
    gemm_wmma<<<ggrid(R_, 256), T, 0, stream>>>(
        envA, W2t, b2, nullptr, 0, A2 + 512, 768,
        R_, 256, 128, 1, nullptr, nullptr);

    // relational conv: scatter-sum then fused [agg|rf] @ [Wconv;Wself] + bconv
    scatter_kernel<<<blocks((size_t)E_), T, 0, stream>>>(edge_src, edge_dst, edge_rel, rfeat, agg);
    packA1_kernel<<<blocks((size_t)R_ * 512), T, 0, stream>>>(agg, rfeat, A1);
    gemm_wmma<<<ggrid(R_, 512), T, 0, stream>>>(
        A1, Wcs, bconv, nullptr, 0, A2, 768,
        R_, 512, 512, 1, nullptr, nullptr);          // hres -> A2[:, 0:512]

    // z = [hres|env] @ We1 + be1
    gemm_wmma<<<ggrid(R_, 512), T, 0, stream>>>(
        A2, We1t, be1, z, 512, nullptr, 0,
        R_, 512, 768, 0, nullptr, nullptr);

    // batch-norm (train stats over R) + relu -> f16
    bn_stats<<<dim3(512), T, 0, stream>>>(z, bnm, bnr);
    bn_apply<<<blocks((size_t)R_ * 512), T, 0, stream>>>(z, bnm, bnr, gamma1, beta1, A3);

    // out = relu(bn(z)) @ We2 + be2  -> d_out f32 [2048,512]
    gemm_wmma<<<ggrid(R_, 512), T, 0, stream>>>(
        A3, We2t, be2, (float*)d_out, 512, nullptr, 0,
        R_, 512, 512, 0, nullptr, nullptr);
}
